// Head_55147380081195
// MI455X (gfx1250) — compile-verified, matrix operand order checked by reading, not run on hardware
//
#include <hip/hip_runtime.h>
#include <hip/hip_bf16.h>

// ---------- problem constants (match reference) ----------
constexpr int Bc = 8;
constexpr int Tc = 2048;
constexpr int DM = 1024;   // d_model
constexpr int DH = 64;     // d_head
constexpr int KC = 32;     // k-chunk for projection GEMM (WMMA K)
constexpr int KT = 32;     // keys per attention tile
constexpr int KTP = 40;    // padded V^T row stride (multiple of 8 -> 16B rows)

typedef __attribute__((ext_vector_type(16))) __bf16 v16bf;
typedef __attribute__((ext_vector_type(8)))  float  v8f;

struct __attribute__((aligned(16))) U128 { unsigned int x, y, z, w; };
struct __attribute__((aligned(16))) F4   { float x, y, z, w; };

union FragU {
    v16bf  v;
    U128   q[2];
    __bf16 e[16];
};

// A-style (and, via transpose-in-LDS, B-style) 16x32 bf16 fragment load.
// lane L: row = L%16; elements 0..7 = k {kb..kb+7}, 8..15 = k {kb+16..kb+23},
// kb = 0 (lanes 0-15) or 8 (lanes 16-31).  Two contiguous 16B loads.
__device__ __forceinline__ v16bf load_frag(const __bf16* p, int stride, int lane) {
    const int r  = lane & 15;
    const int kb = (lane & 16) >> 1;   // 0 or 8
    const __bf16* row = p + r * stride + kb;
    FragU f;
    f.q[0] = *(const U128*)(row);
    f.q[1] = *(const U128*)(row + 16);
    return f.v;
}

__device__ __forceinline__ v8f wmma_bf16(v16bf a, v16bf b, v8f c) {
    return __builtin_amdgcn_wmma_f32_16x16x32_bf16(
        /*neg_a=*/false, a, /*neg_b=*/false, b,
        /*c_mod=*/(short)0, c, /*reuse_a=*/false, /*reuse_b=*/false);
}

// Low 32 bits of a generic pointer to __shared__ == LDS byte offset
// (addrspace(3) -> generic cast places the offset in addr[31:0]).
__device__ __forceinline__ unsigned lds_off(const void* p) {
    return (unsigned)(unsigned long long)p;
}

// =====================================================================
// Kernel 1: Q/K/V projection.  x[B,T,1024] (f32) x W[1024,64] (f32) -> bf16.
// grid (T/128, B), block 256 (8 waves, 16 rows each).
// =====================================================================
__global__ __launch_bounds__(256) void proj_qkv_kernel(
    const float* __restrict__ x,
    const float* __restrict__ wq, const float* __restrict__ wk,
    const float* __restrict__ wv,
    __bf16* __restrict__ Qw, __bf16* __restrict__ Kw, __bf16* __restrict__ Vw)
{
    __shared__ __bf16 Wt[3][DH][KC];   // transposed W chunk: [mat][n][k], 12 KB

    const int tid  = threadIdx.x;
    const int lane = tid & 31;
    const int wave = tid >> 5;
    const int b    = blockIdx.y;
    const int row0 = blockIdx.x * 128 + wave * 16;
    const long xbase = ((long)b * Tc + row0) * DM;

    const float* wmats[3] = {wq, wk, wv};

    v8f acc[3][4];
#pragma unroll
    for (int m = 0; m < 3; ++m)
#pragma unroll
        for (int n = 0; n < 4; ++n) acc[m][n] = (v8f)0.0f;

    const int r  = lane & 15;
    const int kb = (lane & 16) >> 1;

    const int skey = tid >> 3;          // 0..31 : k within chunk
    const int sn0  = (tid & 7) * 8;     // col base for staging

    for (int k0 = 0; k0 < DM; k0 += KC) {
        __syncthreads();
        // ---- cooperative stage of W^T chunks (bf16) ----
#pragma unroll
        for (int m = 0; m < 3; ++m) {
            const F4* src = (const F4*)(wmats[m] + (long)(k0 + skey) * DH + sn0);
            F4 a0 = src[0], a1 = src[1];
            Wt[m][sn0 + 0][skey] = (__bf16)a0.x;
            Wt[m][sn0 + 1][skey] = (__bf16)a0.y;
            Wt[m][sn0 + 2][skey] = (__bf16)a0.z;
            Wt[m][sn0 + 3][skey] = (__bf16)a0.w;
            Wt[m][sn0 + 4][skey] = (__bf16)a1.x;
            Wt[m][sn0 + 5][skey] = (__bf16)a1.y;
            Wt[m][sn0 + 6][skey] = (__bf16)a1.z;
            Wt[m][sn0 + 7][skey] = (__bf16)a1.w;
        }
        __syncthreads();

        // ---- per-wave x A-fragment (f32 -> bf16) ----
        FragU xf;
        const float* xr = x + xbase + (long)r * DM + k0 + kb;
        F4 c0 = *(const F4*)(xr);
        F4 c1 = *(const F4*)(xr + 4);
        F4 c2 = *(const F4*)(xr + 16);
        F4 c3 = *(const F4*)(xr + 20);
        xf.e[0]  = (__bf16)c0.x; xf.e[1]  = (__bf16)c0.y;
        xf.e[2]  = (__bf16)c0.z; xf.e[3]  = (__bf16)c0.w;
        xf.e[4]  = (__bf16)c1.x; xf.e[5]  = (__bf16)c1.y;
        xf.e[6]  = (__bf16)c1.z; xf.e[7]  = (__bf16)c1.w;
        xf.e[8]  = (__bf16)c2.x; xf.e[9]  = (__bf16)c2.y;
        xf.e[10] = (__bf16)c2.z; xf.e[11] = (__bf16)c2.w;
        xf.e[12] = (__bf16)c3.x; xf.e[13] = (__bf16)c3.y;
        xf.e[14] = (__bf16)c3.z; xf.e[15] = (__bf16)c3.w;

#pragma unroll
        for (int m = 0; m < 3; ++m)
#pragma unroll
            for (int n = 0; n < 4; ++n) {
                v16bf bf = load_frag(&Wt[m][n * 16][0], KC, lane);
                acc[m][n] = wmma_bf16(xf.v, bf, acc[m][n]);
            }
    }

    // ---- write out bf16 Q (scaled by 1/sqrt(DH)), K, V ----
    __bf16* outs[3] = {Qw, Kw, Vw};
    const float scl[3] = {0.125f, 1.0f, 1.0f};
    const int half = (lane >> 4) & 1;
    const int nl   = lane & 15;
#pragma unroll
    for (int m = 0; m < 3; ++m)
#pragma unroll
        for (int n = 0; n < 4; ++n)
#pragma unroll
            for (int rr = 0; rr < 8; ++rr) {
                const int mrow = rr + half * 8;
                const int col  = n * 16 + nl;
                outs[m][((long)b * Tc + row0 + mrow) * DH + col] =
                    (__bf16)(acc[m][n][rr] * scl[m]);
            }
}

// =====================================================================
// Kernel 2: causal flash attention over bf16 Q/K/V -> f32 out.
// grid (T/128, B), block 256 (8 waves, 16 q-rows each).
// K tile staged via CDNA5 async global->LDS DMA (ASYNCcnt); V staged
// transposed through VGPRs; next tile prefetched into L2.
// =====================================================================
__global__ __launch_bounds__(256) void flash_attn_kernel(
    const __bf16* __restrict__ Qw, const __bf16* __restrict__ Kw,
    const __bf16* __restrict__ Vw, float* __restrict__ out)
{
    __shared__ __bf16 Kt[KT][DH];        // 4 KB  (row-major keys)
    __shared__ __bf16 Vt[DH][KTP];       // 5 KB  (V transposed)
    __shared__ __bf16 Ps[8][16][KT];     // 8 KB  (per-wave P bounce)

    const int tid  = threadIdx.x;
    const int lane = tid & 31;
    const int wave = tid >> 5;
    const int b    = blockIdx.y;
    const int q0   = blockIdx.x * 128 + wave * 16;
    const int half = (lane >> 4) & 1;
    const int nl   = lane & 15;

    // Q fragments (K-dim 0..31 and 32..63) straight from global bf16.
    const __bf16* qbase = Qw + ((long)b * Tc + q0) * DH;
    const v16bf qf0 = load_frag(qbase,      DH, lane);
    const v16bf qf1 = load_frag(qbase + 32, DH, lane);

    v8f oacc[4];
#pragma unroll
    for (int n = 0; n < 4; ++n) oacc[n] = (v8f)0.0f;
    float mi[8], li[8];
#pragma unroll
    for (int rr = 0; rr < 8; ++rr) { mi[rr] = -1e30f; li[rr] = 0.0f; }

    const int skey = tid >> 3;
    const int sd0  = (tid & 7) * 8;
    const int kend = blockIdx.x * 128 + 128;   // causal bound for this WG
    const unsigned kt_lds = lds_off(&Kt[skey][sd0]);

    for (int kb0 = 0; kb0 < kend; kb0 += KT) {
        __syncthreads();
        // ---- stage K via async global->LDS DMA; V transposed via VGPRs ----
        {
            const long grow = (long)b * Tc + kb0 + skey;
            const unsigned long long kgaddr =
                (unsigned long long)(Kw + grow * DH + sd0);
            asm volatile("global_load_async_to_lds_b128 %0, %1, off"
                         :: "v"(kt_lds), "v"(kgaddr) : "memory");

            U128 vv = *(const U128*)(Vw + grow * DH + sd0);
            const __bf16* ve = (const __bf16*)&vv;
#pragma unroll
            for (int j = 0; j < 8; ++j) Vt[sd0 + j][skey] = ve[j];

            // L2 prefetch of next tile while this one computes
            if (kb0 + KT < kend) {
                __builtin_prefetch(Kw + (grow + KT) * DH + sd0, 0, 3);
                __builtin_prefetch(Vw + (grow + KT) * DH + sd0, 0, 3);
            }
        }
        asm volatile("s_wait_asynccnt 0x0" ::: "memory");
        __syncthreads();

        if (kb0 < q0 + 16) {   // wave-uniform: tile intersects this wave's rows
            // ---- S = Q K^T  (two 16x16 tiles, K accumulated over 64) ----
            v8f s[2];
#pragma unroll
            for (int nc = 0; nc < 2; ++nc) {
                v16bf kf0 = load_frag(&Kt[nc * 16][0],      DH, lane);
                v16bf kf1 = load_frag(&Kt[nc * 16][0] + 32, DH, lane);
                s[nc] = (v8f)0.0f;
                s[nc] = wmma_bf16(qf0, kf0, s[nc]);
                s[nc] = wmma_bf16(qf1, kf1, s[nc]);
            }
            // ---- causal mask + online softmax (C-layout native) ----
#pragma unroll
            for (int rr = 0; rr < 8; ++rr) {
                const int mrow = q0 + rr + half * 8;
                float a0 = s[0][rr];
                float a1 = s[1][rr];
                if (kb0 + nl      > mrow) a0 = -1e30f;
                if (kb0 + 16 + nl > mrow) a1 = -1e30f;
                float mx = fmaxf(a0, a1);
                mx = fmaxf(mx, __shfl_xor(mx, 1, 32));
                mx = fmaxf(mx, __shfl_xor(mx, 2, 32));
                mx = fmaxf(mx, __shfl_xor(mx, 4, 32));
                mx = fmaxf(mx, __shfl_xor(mx, 8, 32));
                const float mnew  = fmaxf(mi[rr], mx);
                const float scale = __expf(mi[rr] - mnew);
                const float p0 = __expf(a0 - mnew);
                const float p1 = __expf(a1 - mnew);
                Ps[wave][rr + half * 8][nl]      = (__bf16)p0;
                Ps[wave][rr + half * 8][16 + nl] = (__bf16)p1;
                float sum = p0 + p1;
                sum += __shfl_xor(sum, 1, 32);
                sum += __shfl_xor(sum, 2, 32);
                sum += __shfl_xor(sum, 4, 32);
                sum += __shfl_xor(sum, 8, 32);
                li[rr] = li[rr] * scale + sum;
                mi[rr] = mnew;
#pragma unroll
                for (int n = 0; n < 4; ++n) oacc[n][rr] *= scale;
            }
            // P (C-layout) was bounced through LDS; wait for DS, reload A-layout.
            asm volatile("s_wait_dscnt 0" ::: "memory");
            const v16bf pf = load_frag(&Ps[wave][0][0], KT, lane);
            // ---- O += P V ----
#pragma unroll
            for (int n = 0; n < 4; ++n) {
                v16bf vf = load_frag(&Vt[n * 16][0], KTP, lane);
                oacc[n] = wmma_bf16(pf, vf, oacc[n]);
            }
        }
    }

    // ---- epilogue: normalize, store f32 ----
#pragma unroll
    for (int n = 0; n < 4; ++n)
#pragma unroll
        for (int rr = 0; rr < 8; ++rr) {
            const int mrow = q0 + rr + half * 8;
            const int col  = n * 16 + nl;
            out[((long)b * Tc + mrow) * DH + col] = oacc[n][rr] / li[rr];
        }
}

// =====================================================================
extern "C" void kernel_launch(void* const* d_in, const int* in_sizes, int n_in,
                              void* d_out, int out_size, void* d_ws, size_t ws_size,
                              hipStream_t stream) {
    const float* x  = (const float*)d_in[0];
    const float* wq = (const float*)d_in[1];
    const float* wk = (const float*)d_in[2];
    const float* wv = (const float*)d_in[3];
    float* out = (float*)d_out;

    const size_t per = (size_t)Bc * Tc * DH;   // elements per bf16 matrix
    __bf16* Qw = (__bf16*)d_ws;
    __bf16* Kw = Qw + per;
    __bf16* Vw = Kw + per;

    dim3 grid(Tc / 128, Bc);
    dim3 block(256);
    proj_qkv_kernel<<<grid, block, 0, stream>>>(x, wq, wk, wv, Qw, Kw, Vw);
    flash_attn_kernel<<<grid, block, 0, stream>>>(Qw, Kw, Vw, out);
}
